// StructureAwareGATCoxModel_15796889714827
// MI455X (gfx1250) — matrix operand order repaired and checked by reading, not compiled, not run on hardware
//
#include <hip/hip_runtime.h>

#define DEV __device__ __forceinline__

namespace {

// ---------------- problem constants ----------------
constexpr int G_     = 64;
constexpr int NPG_   = 256;
constexpr int NNODES = G_ * NPG_;        // 16384
constexpr int EPG_   = 4096;
constexpr int NEDGES = G_ * EPG_;        // 262144
constexpr int FN_    = 128;
constexpr int HID_   = 64;
constexpr int HEADS_ = 4;
constexpr int EDGEH_ = 12;
constexpr int HC_    = HID_ * HEADS_;    // 256
constexpr int ETOT   = NEDGES + NNODES;  // 278528 (edges + self loops)

// ---------------- output layout (floats) ----------------
constexpr int OUT_EMB = 64;                       // risk[64] first
constexpr int OUT_LAT = OUT_EMB + G_ * 192;       // 12352
constexpr int OUT_AUX = OUT_LAT + G_ * 32;        // 14400
constexpr int OUT_GT  = OUT_AUX + 1;              // 14401

// ---------------- workspace layout (bytes) ----------------
constexpr size_t OFS_ADJ  = 0;
constexpr size_t OFS_WADJ = OFS_ADJ  + (size_t)G_*NPG_*NPG_*4;
constexpr size_t OFS_ADJH = OFS_WADJ + (size_t)G_*NPG_*NPG_*4;   // bf16 (ushort)
constexpr size_t OFS_HX   = OFS_ADJH + (size_t)G_*NPG_*NPG_*2;
constexpr size_t OFS_ACC  = OFS_HX   + (size_t)NNODES*HC_*4;
constexpr size_t OFS_X0H  = OFS_ACC  + (size_t)NNODES*HC_*4;     // bf16
constexpr size_t OFS_H1H  = OFS_X0H  + (size_t)NNODES*HID_*2;    // bf16
constexpr size_t OFS_H2H  = OFS_H1H  + (size_t)NNODES*HC_*2;     // bf16
constexpr size_t OFS_H3   = OFS_H2H  + (size_t)NNODES*HC_*2;     // f32
constexpr size_t OFS_EAA  = OFS_H3   + (size_t)NNODES*HID_*4;
constexpr size_t OFS_ASC  = OFS_EAA  + (size_t)ETOT*EDGEH_*4;
constexpr size_t OFS_AMAX = OFS_ASC  + (size_t)ETOT*HEADS_*4;
constexpr size_t OFS_ASUM = OFS_AMAX + (size_t)NNODES*HEADS_*4;
constexpr size_t OFS_ASRC = OFS_ASUM + (size_t)NNODES*HEADS_*4;
constexpr size_t OFS_ADST = OFS_ASRC + (size_t)NNODES*HEADS_*4;
constexpr size_t OFS_CNT  = OFS_ADST + (size_t)NNODES*HEADS_*4;
constexpr size_t OFS_LOOP = OFS_CNT  + (size_t)NNODES*4;
constexpr size_t OFS_DEG  = OFS_LOOP + (size_t)NNODES*EDGEH_*4;
constexpr size_t OFS_WDEG = OFS_DEG  + (size_t)NNODES*4;
constexpr size_t OFS_TRI  = OFS_WDEG + (size_t)NNODES*4;
constexpr size_t OFS_LDEG = OFS_TRI  + (size_t)NNODES*4;
constexpr size_t OFS_WROW = OFS_LDEG + (size_t)NNODES*4;
constexpr size_t OFS_CLU  = OFS_WROW + (size_t)NNODES*4;
constexpr size_t OFS_TSC  = OFS_CLU  + (size_t)NNODES*4;
constexpr size_t OFS_BRP  = OFS_TSC  + (size_t)NNODES*4;
constexpr size_t OFS_MAXV = OFS_BRP  + (size_t)NNODES*4;         // 3 uints
constexpr size_t OFS_WEFF = OFS_MAXV + 256;
constexpr size_t OFS_W1T  = OFS_WEFF + 256;                      // bf16 weight^T
constexpr size_t OFS_W2T  = OFS_W1T + (size_t)HC_*HID_*2;
constexpr size_t OFS_W3T  = OFS_W2T + (size_t)HC_*HC_*2;
constexpr size_t OFS_R1T  = OFS_W3T + (size_t)HID_*HC_*2;
constexpr size_t OFS_R2T  = OFS_R1T + (size_t)HC_*HID_*2;
constexpr size_t OFS_R3T  = OFS_R2T + (size_t)HC_*HC_*2;

// ---------------- types for WMMA ----------------
typedef __bf16 bf16_t;
typedef bf16_t  v16bf  __attribute__((ext_vector_type(16)));
typedef float   v8f    __attribute__((ext_vector_type(8)));
typedef unsigned uint4v __attribute__((ext_vector_type(4)));

union FragU { v16bf v; uint4v q[2]; };

// ---------------- helpers ----------------
DEV float bf2f(unsigned short h) { return __uint_as_float(((unsigned)h) << 16); }
DEV unsigned short f2bf(float f) {
  unsigned u = __float_as_uint(f);
  u += 0x7fffu + ((u >> 16) & 1u);      // round-to-nearest-even
  return (unsigned short)(u >> 16);
}
// monotonic float<->uint encoding for signed atomic max
DEV unsigned encf(float f) {
  unsigned u = __float_as_uint(f);
  return (u & 0x80000000u) ? ~u : (u | 0x80000000u);
}
DEV float decf(unsigned u) {
  return __uint_as_float((u & 0x80000000u) ? (u & 0x7fffffffu) : ~u);
}

// one 16x16x32 bf16 WMMA step; A row-major MxK, Bt row-major NxK (B transposed)
DEV v8f wmma_step(const unsigned short* __restrict__ arow,
                  const unsigned short* __restrict__ btrow,
                  int kb, int h, v8f acc) {
  FragU fa, fb;
  fa.q[0] = *(const uint4v*)(arow + kb + h * 8);
  fa.q[1] = *(const uint4v*)(arow + kb + 16 + h * 8);
  fb.q[0] = *(const uint4v*)(btrow + kb + h * 8);
  fb.q[1] = *(const uint4v*)(btrow + kb + 16 + h * 8);
  return __builtin_amdgcn_wmma_f32_16x16x32_bf16(false, fa.v, false, fb.v,
                                                 (short)0, acc, false, false);
}

// ---------------- generic WMMA GEMM: C = A(MxK) * Bt^T + bias ----------------
__global__ void k_gemm(const unsigned short* __restrict__ A,
                       const unsigned short* __restrict__ Bt,
                       const float* __restrict__ bias,
                       float* __restrict__ C, int M, int Nc, int K, int relu) {
  int lane = threadIdx.x & 31, wave = threadIdx.x >> 5;
  int m0 = (blockIdx.y * 4 + wave) * 16;
  int n0 = blockIdx.x * 16;
  if (m0 >= M) return;                  // wave-uniform
  int h = lane >> 4, c = lane & 15;
  const unsigned short* arow = A  + (size_t)(m0 + c) * K;
  const unsigned short* brow = Bt + (size_t)(n0 + c) * K;
  v8f acc = {0.f, 0.f, 0.f, 0.f, 0.f, 0.f, 0.f, 0.f};
  for (int kb = 0; kb < K; kb += 32) acc = wmma_step(arow, brow, kb, h, acc);
#pragma unroll
  for (int r = 0; r < 8; ++r) {
    int row = m0 + r + 8 * h, col = n0 + c;
    float v = acc[r] + (bias ? bias[col] : 0.f);
    if (relu) v = fmaxf(v, 0.f);
    C[(size_t)row * Nc + col] = v;
  }
}

// ---------------- fused adj@adj + triangle count ----------------
// tri[g,i] += sum_j (adj@adj)[i,j] * adj[i,j]   (adj symmetric bf16)
__global__ void k_adjtri(const unsigned short* __restrict__ adjh,
                         float* __restrict__ tri) {
  int g = blockIdx.z;
  const unsigned short* Ag = adjh + ((size_t)g << 16);
  int lane = threadIdx.x & 31, wave = threadIdx.x >> 5;
  int m0 = (blockIdx.y * 4 + wave) * 16;
  int n0 = blockIdx.x * 16;
  int h = lane >> 4, c = lane & 15;
  const unsigned short* arow = Ag + (size_t)(m0 + c) * NPG_;
  const unsigned short* brow = Ag + (size_t)(n0 + c) * NPG_;  // symmetric: col == row
  v8f acc = {0.f, 0.f, 0.f, 0.f, 0.f, 0.f, 0.f, 0.f};
  for (int kb = 0; kb < NPG_; kb += 32) acc = wmma_step(arow, brow, kb, h, acc);
#pragma unroll
  for (int r = 0; r < 8; ++r) {
    int row = m0 + r + 8 * h;
    float v = acc[r] * bf2f(Ag[(size_t)row * NPG_ + n0 + c]);
    v += __shfl_xor(v, 1, 16);
    v += __shfl_xor(v, 2, 16);
    v += __shfl_xor(v, 4, 16);
    v += __shfl_xor(v, 8, 16);
    if (c == 0) atomicAdd(&tri[g * NPG_ + row], v);
  }
}

// ---------------- small utility kernels ----------------
__global__ void k_zero_f(float* p, long long n) {
  long long i = (long long)blockIdx.x * 256 + threadIdx.x;
  if (i < n) p[i] = 0.f;
}
__global__ void k_zero_u(unsigned* p, long long n) {
  long long i = (long long)blockIdx.x * 256 + threadIdx.x;
  if (i < n) p[i] = 0u;
}
__global__ void k_wt(const float* __restrict__ W, unsigned short* __restrict__ Wt,
                     int K, int Nc) {
  int idx = blockIdx.x * 256 + threadIdx.x;
  if (idx >= K * Nc) return;
  int k = idx / Nc, n = idx - k * Nc;
  Wt[(size_t)n * K + k] = f2bf(W[idx]);
}

// ---------------- structure features ----------------
__global__ void k_deg(const int* __restrict__ src, const float* __restrict__ ew,
                      float* deg, float* wdeg) {
  int e = blockIdx.x * 256 + threadIdx.x;
  if (e >= NEDGES) return;
  atomicAdd(&deg[src[e]], 1.f);
  atomicAdd(&wdeg[src[e]], ew[e]);
}
__global__ void k_adjbuild(const int* __restrict__ src, const int* __restrict__ dst,
                           const float* __restrict__ ew, float* adj, float* wadj) {
  int e = blockIdx.x * 256 + threadIdx.x;
  if (e >= NEDGES) return;
  int s = src[e], d = dst[e];
  int g = s >> 8, ls = s & 255, ld = d & 255;
  if (ls != ld) {
    size_t base = ((size_t)g << 16) + (size_t)ls * NPG_ + ld;
    atomicMax((int*)(adj + base), __float_as_int(1.0f));
    atomicMax((int*)(wadj + base), __float_as_int(ew[e]));
  }
}
__global__ void k_adjh(const float* __restrict__ adj, unsigned short* __restrict__ adjh) {
  int idx = blockIdx.x * 256 + threadIdx.x;
  if (idx >= G_ * NPG_ * NPG_) return;
  int g = idx >> 16, rem = idx & 65535, i = rem >> 8, j = rem & 255;
  float v = fmaxf(adj[idx], adj[((size_t)g << 16) + (size_t)j * NPG_ + i]);
  adjh[idx] = f2bf(v);
}
__global__ void k_rowsum(const unsigned short* __restrict__ adjh,
                         const float* __restrict__ wadj, float* ldeg, float* wrow) {
  int g = blockIdx.x, i = threadIdx.x;
  const unsigned short* ar = adjh + ((size_t)g << 16) + (size_t)i * NPG_;
  const float* wg = wadj + ((size_t)g << 16);
  float ls = 0.f, ws = 0.f;
  for (int j = 0; j < NPG_; ++j) {
    ls += bf2f(ar[j]);
    ws += fmaxf(wg[(size_t)i * NPG_ + j], wg[(size_t)j * NPG_ + i]);
  }
  ldeg[g * NPG_ + i] = ls;
  wrow[g * NPG_ + i] = ws;
}
__global__ void k_maxdw(const float* deg, const float* wdeg, unsigned* maxv) {
  int i = blockIdx.x * 256 + threadIdx.x;
  if (i >= NNODES) return;
  atomicMax(&maxv[0], __float_as_uint(deg[i]));   // nonneg
  atomicMax(&maxv[1], __float_as_uint(wdeg[i]));
}
__global__ void k_structfin(const float* __restrict__ ldeg, const float* __restrict__ tri,
                            const float* __restrict__ wrow, float* clu, float* tsc,
                            float* brp, unsigned* maxv, float* out) {
  __shared__ float sr[NPG_];
  int g = blockIdx.x, t = threadIdx.x;
  int i = g * NPG_ + t;
  float ld = ldeg[i];
  float tv = 0.5f * tri[i];
  float cl = tv / fmaxf(ld * (ld - 1.f) * 0.5f, 1.f);
  float br = ld * (1.f - cl);
  clu[i] = cl;
  tsc[i] = tv / (float)NPG_;
  float bp = br / (float)NPG_;
  brp[i] = bp;
  atomicMax(&maxv[2], __float_as_uint(bp));       // nonneg
  sr[t] = cl; __syncthreads();
  for (int s = 128; s > 0; s >>= 1) { if (t < s) sr[t] += sr[t + s]; __syncthreads(); }
  float clm = sr[0] / (float)NPG_; __syncthreads();
  sr[t] = wrow[i]; __syncthreads();
  for (int s = 128; s > 0; s >>= 1) { if (t < s) sr[t] += sr[t + s]; __syncthreads(); }
  float wm = sr[0] / (float)NPG_; __syncthreads();
  sr[t] = br; __syncthreads();
  for (int s = 128; s > 0; s >>= 1) { if (t < s) sr[t] += sr[t + s]; __syncthreads(); }
  float brm = sr[0] / (float)NPG_;
  if (t == 0)
    out[OUT_GT + g] = 0.45f * clm + 0.35f * (wm / (float)NPG_) + 0.2f * brm;
}

// ---------------- node projection (K=133, VALU) ----------------
__global__ void k_nodeproj(const float* __restrict__ x, const float* deg, const float* wdeg,
                           const float* clu, const float* tsc, const float* brp,
                           const unsigned* maxv, const float* __restrict__ w,
                           const float* __restrict__ b, unsigned short* x0h) {
  int idx = blockIdx.x * 256 + threadIdx.x;
  if (idx >= NNODES * HID_) return;
  int n = idx / HID_, o = idx - n * HID_;
  float dm = fmaxf(__uint_as_float(maxv[0]), 1.f);
  float wm = fmaxf(__uint_as_float(maxv[1]), 1.f);
  float bm = fmaxf(__uint_as_float(maxv[2]), 1.f);
  const float* xr = x + (size_t)n * FN_;
  float a = b[o];
  for (int k = 0; k < FN_; ++k) a += xr[k] * w[k * HID_ + o];
  a += (deg[n] / dm)  * w[128 * HID_ + o];
  a += (wdeg[n] / wm) * w[129 * HID_ + o];
  a += clu[n]         * w[130 * HID_ + o];
  a += tsc[n]         * w[131 * HID_ + o];
  a += (brp[n] / bm)  * w[132 * HID_ + o];
  x0h[idx] = f2bf(fmaxf(a, 0.f));
}

// ---------------- edge encoder + self-loop features ----------------
__global__ void k_edgeenc(const float* __restrict__ eattr, const float* w1, const float* b1,
                          const float* w2, const float* b2, float* __restrict__ eaa) {
  int e = blockIdx.x * 256 + threadIdx.x;
  if (e >= NEDGES) return;
  float v = eattr[e];
  float t1[EDGEH_];
  for (int j = 0; j < EDGEH_; ++j) t1[j] = fmaxf(v * w1[j] + b1[j], 0.f);
  for (int o = 0; o < EDGEH_; ++o) {
    float a = b2[o];
    for (int j = 0; j < EDGEH_; ++j) a += t1[j] * w2[j * EDGEH_ + o];
    eaa[(size_t)e * EDGEH_ + o] = fmaxf(a, 0.f);
  }
}
__global__ void k_loopacc(const int* __restrict__ dst, const float* __restrict__ eaa,
                          float* cnt, float* loop) {
  int e = blockIdx.x * 256 + threadIdx.x;
  if (e >= NEDGES) return;
  int d = dst[e];
  atomicAdd(&cnt[d], 1.f);
  for (int j = 0; j < EDGEH_; ++j)
    atomicAdd(&loop[(size_t)d * EDGEH_ + j], eaa[(size_t)e * EDGEH_ + j]);
}
__global__ void k_loopwrite(const float* cnt, const float* loop, float* eaa) {
  int idx = blockIdx.x * 256 + threadIdx.x;
  if (idx >= NNODES * EDGEH_) return;
  int n = idx / EDGEH_;
  eaa[(size_t)(NEDGES + n) * EDGEH_ + (idx - n * EDGEH_)] =
      loop[idx] / fmaxf(cnt[n], 1.f);
}

// ---------------- GAT attention ----------------
// weff[k,h] = sum_c lin_edge_w[k, h*64+c] * att_edge[h,c]
__global__ void k_weff(const float* __restrict__ lew, const float* __restrict__ ae,
                       float* weff, int H) {
  int idx = threadIdx.x;
  if (idx >= EDGEH_ * H) return;
  int k = idx / H, h = idx - k * H;
  float s = 0.f;
  for (int c = 0; c < 64; ++c) s += lew[(size_t)k * H * 64 + h * 64 + c] * ae[h * 64 + c];
  weff[idx] = s;
}
__global__ void k_srcdst(const float* __restrict__ hx, const float* as, const float* ad,
                         float* asrc, float* adst, int H) {
  int idx = blockIdx.x * 256 + threadIdx.x;
  if (idx >= NNODES * H) return;
  int n = idx / H, h = idx - n * H;
  const float* row = hx + (size_t)n * H * 64 + h * 64;
  float s = 0.f, d = 0.f;
  for (int c = 0; c < 64; ++c) { s += row[c] * as[h * 64 + c]; d += row[c] * ad[h * 64 + c]; }
  asrc[idx] = s;
  adst[idx] = d;
}
__global__ void k_attA(const int* __restrict__ src, const int* __restrict__ dst,
                       const float* __restrict__ eaa, const float* __restrict__ weff,
                       const float* __restrict__ asrc, const float* __restrict__ adst,
                       float* asc, unsigned* amax, int H) {
  int idx = blockIdx.x * 256 + threadIdx.x;
  if (idx >= ETOT * H) return;
  int e = idx / H, h = idx - e * H;
  int s = e < NEDGES ? src[e] : (e - NEDGES);
  int d = e < NEDGES ? dst[e] : (e - NEDGES);
  float a = asrc[s * H + h] + adst[d * H + h];
  const float* er = eaa + (size_t)e * EDGEH_;
  for (int j = 0; j < EDGEH_; ++j) a += er[j] * weff[j * H + h];
  a = a > 0.f ? a : 0.2f * a;           // leaky_relu(0.2)
  asc[idx] = a;
  atomicMax(&amax[d * H + h], encf(a));
}
__global__ void k_attB(const int* __restrict__ dst, float* asc,
                       const unsigned* __restrict__ amax, float* asum, int H) {
  int idx = blockIdx.x * 256 + threadIdx.x;
  if (idx >= ETOT * H) return;
  int e = idx / H, h = idx - e * H;
  int d = e < NEDGES ? dst[e] : (e - NEDGES);
  float ex = __expf(asc[idx] - decf(amax[d * H + h]));
  asc[idx] = ex;
  atomicAdd(&asum[d * H + h], ex);
}
__global__ void k_attC(const int* __restrict__ src, const int* __restrict__ dst,
                       const float* __restrict__ asc, const float* __restrict__ asum,
                       const float* __restrict__ hx, float* accb, int H) {
  int idx = blockIdx.x * 256 + threadIdx.x;
  if (idx >= ETOT * H) return;
  int e = idx / H, h = idx - e * H;
  int s = e < NEDGES ? src[e] : (e - NEDGES);
  int d = e < NEDGES ? dst[e] : (e - NEDGES);
  float coef = asc[idx] / (asum[d * H + h] + 1e-16f);
  int oc = H * 64;
  const float* hr = hx + (size_t)s * oc + h * 64;
  float* ar = accb + (size_t)d * oc + h * 64;
  for (int c = 0; c < 64; ++c) atomicAdd(&ar[c], hr[c] * coef);
}

// ---------------- LayerNorm (+gat bias) + ReLU, wave per row ----------------
__global__ void k_ln(const float* __restrict__ acc, const float* __restrict__ gbias,
                     const float* __restrict__ lg, const float* __restrict__ lb,
                     unsigned short* obf, float* of32, int W, int rows) {
  int wid = threadIdx.x >> 5, lane = threadIdx.x & 31;
  int row = blockIdx.x * (blockDim.x >> 5) + wid;
  if (row >= rows) return;
  int per = W >> 5;                      // 8 (W=256) or 2 (W=64)
  const float* r = acc + (size_t)row * W;
  float v[8];
  float s = 0.f;
  for (int i = 0; i < per; ++i) { int c = lane * per + i; v[i] = r[c] + gbias[c]; s += v[i]; }
  for (int o = 16; o > 0; o >>= 1) s += __shfl_xor(s, o, 32);
  float mu = s / (float)W;
  float q = 0.f;
  for (int i = 0; i < per; ++i) { float d = v[i] - mu; q += d * d; }
  for (int o = 16; o > 0; o >>= 1) q += __shfl_xor(q, o, 32);
  float inv = rsqrtf(q / (float)W + 1e-5f);
  for (int i = 0; i < per; ++i) {
    int c = lane * per + i;
    float ov = fmaxf((v[i] - mu) * inv * lg[c] + lb[c], 0.f);
    if (obf)  obf[(size_t)row * W + c] = f2bf(ov);
    if (of32) of32[(size_t)row * W + c] = ov;
  }
}

// ---------------- pooling: mean / max / attention, per graph ----------------
__global__ void k_pool(const float* __restrict__ h3, const float* __restrict__ g1w,
                       const float* __restrict__ g1b, const float* __restrict__ g2w,
                       const float* __restrict__ g2b, float* __restrict__ out) {
  __shared__ float she[NPG_];
  __shared__ float shr[NPG_];
  int g = blockIdx.x, t = threadIdx.x;
  const float* hrow = h3 + ((size_t)g * NPG_ + t) * HID_;
  float hr[HID_];
  for (int k = 0; k < HID_; ++k) hr[k] = hrow[k];
  float gate = g2b[0];
  for (int j = 0; j < 32; ++j) {
    float a = g1b[j];
    for (int k = 0; k < HID_; ++k) a += hr[k] * g1w[k * 32 + j];
    gate += fmaxf(a, 0.f) * g2w[j];
  }
  shr[t] = gate; __syncthreads();
  for (int s = 128; s > 0; s >>= 1) { if (t < s) shr[t] = fmaxf(shr[t], shr[t + s]); __syncthreads(); }
  float gmax = shr[0]; __syncthreads();
  float ex = __expf(gate - gmax);
  she[t] = ex; shr[t] = ex; __syncthreads();
  for (int s = 128; s > 0; s >>= 1) { if (t < s) shr[t] += shr[t + s]; __syncthreads(); }
  float denom = shr[0] + 1e-16f;
  __syncthreads();
  if (t < HID_) {
    float sm = 0.f, mx = -3.4e38f, ap = 0.f;
    for (int n = 0; n < NPG_; ++n) {
      float v = h3[((size_t)g * NPG_ + n) * HID_ + t];
      sm += v;
      mx = fmaxf(mx, v);
      ap += (she[n] / denom) * v;
    }
    out[OUT_EMB + g * 192 + t]        = sm / (float)NPG_;
    out[OUT_EMB + g * 192 + 64 + t]   = mx;
    out[OUT_EMB + g * 192 + 128 + t]  = ap;
  }
}

// ---------------- fusion head: fus1 -> fus2 -> risk ----------------
__global__ void k_fus(const float* __restrict__ clin, const float* __restrict__ met,
                      const float* f1w, const float* f1b, const float* f2w,
                      const float* f2b, const float* rw, const float* rb,
                      float* __restrict__ out) {
  __shared__ float fx[352];
  __shared__ float l1[64];
  __shared__ float l2[32];
  int g = blockIdx.x, t = threadIdx.x;  // 64 threads
  for (int i = t; i < 352; i += 64) {
    float v;
    if (i < 192)      v = out[OUT_EMB + g * 192 + i];
    else if (i < 224) v = clin[g * 32 + (i - 192)];
    else              v = met[g * 128 + (i - 224)];
    fx[i] = v;
  }
  __syncthreads();
  {
    float a = f1b[t];
    for (int k = 0; k < 352; ++k) a += fx[k] * f1w[k * 64 + t];
    l1[t] = fmaxf(a, 0.f);
  }
  __syncthreads();
  if (t < 32) {
    float a = f2b[t];
    for (int k = 0; k < 64; ++k) a += l1[k] * f2w[k * 32 + t];
    float v = fmaxf(a, 0.f);
    l2[t] = v;
    out[OUT_LAT + g * 32 + t] = v;
  }
  __syncthreads();
  if (t == 0) {
    float a = rb[0];
    for (int k = 0; k < 32; ++k) a += l2[k] * rw[k];
    out[g] = a;
  }
}

// ---------------- structure head: sh1 -> sh2 -> aux MSE ----------------
__global__ void k_shaux(const float* s1w, const float* s1b, const float* s2w,
                        const float* s2b, float* __restrict__ out) {
  __shared__ float s1[32];
  int g = blockIdx.x, t = threadIdx.x;  // 32 threads
  float a = s1b[t];
  for (int k = 0; k < 192; ++k) a += out[OUT_EMB + g * 192 + k] * s1w[k * 32 + t];
  s1[t] = fmaxf(a, 0.f);
  __syncthreads();
  if (t == 0) {
    float sp = s2b[0];
    for (int k = 0; k < 32; ++k) sp += s1[k] * s2w[k];
    float d = sp - out[OUT_GT + g];
    atomicAdd(&out[OUT_AUX], d * d * (1.0f / (float)G_));
  }
}

inline unsigned gcnt(long long n) { return (unsigned)((n + 255) / 256); }

} // namespace

extern "C" void kernel_launch(void* const* d_in, const int* in_sizes, int n_in,
                              void* d_out, int out_size, void* d_ws, size_t ws_size,
                              hipStream_t stream) {
  (void)in_sizes; (void)n_in; (void)out_size; (void)ws_size;
  const float* X     = (const float*)d_in[0];
  const int*   EIDX  = (const int*)d_in[1];
  const float* EATTR = (const float*)d_in[2];
  const float* CLINI = (const float*)d_in[4];
  const float* METAB = (const float*)d_in[5];
  const int* src = EIDX;
  const int* dst = EIDX + NEDGES;
  auto P = [&](int i) { return (const float*)d_in[i]; };
  // param leaf order (jax pytree: dict keys sorted):
  // 6 ee1.b 7 ee1.w 8 ee2.b 9 ee2.w 10 fus1.b 11 fus1.w 12 fus2.b 13 fus2.w
  // 14-19 gat1{att_dst,att_edge,att_src,bias,lin_edge_w,lin_w}
  // 20-25 gat2{...} 26-31 gat3{...}
  // 32 gate1.b 33 gate1.w 34 gate2.b 35 gate2.w 36 np.b 37 np.w
  // 38 n1.b 39 n1.g 40 n2.b 41 n2.g 42 n3.b 43 n3.g
  // 44 res1.b 45 res1.w 46 res2.b 47 res2.w 48 res3.b 49 res3.w
  // 50 risk.b 51 risk.w 52 sh1.b 53 sh1.w 54 sh2.b 55 sh2.w

  char* ws = (char*)d_ws;
  float* ADJ  = (float*)(ws + OFS_ADJ);
  float* WADJ = (float*)(ws + OFS_WADJ);
  unsigned short* ADJH = (unsigned short*)(ws + OFS_ADJH);
  float* HX  = (float*)(ws + OFS_HX);
  float* ACC = (float*)(ws + OFS_ACC);
  unsigned short* X0H = (unsigned short*)(ws + OFS_X0H);
  unsigned short* H1H = (unsigned short*)(ws + OFS_H1H);
  unsigned short* H2H = (unsigned short*)(ws + OFS_H2H);
  float* H3   = (float*)(ws + OFS_H3);
  float* EAA  = (float*)(ws + OFS_EAA);
  float* ASC  = (float*)(ws + OFS_ASC);
  unsigned* AMAX = (unsigned*)(ws + OFS_AMAX);
  float* ASUM = (float*)(ws + OFS_ASUM);
  float* ASRC = (float*)(ws + OFS_ASRC);
  float* ADST = (float*)(ws + OFS_ADST);
  float* CNT  = (float*)(ws + OFS_CNT);
  float* LOOP = (float*)(ws + OFS_LOOP);
  float* DEG  = (float*)(ws + OFS_DEG);
  float* WDEG = (float*)(ws + OFS_WDEG);
  float* TRI  = (float*)(ws + OFS_TRI);
  float* LDEG = (float*)(ws + OFS_LDEG);
  float* WROW = (float*)(ws + OFS_WROW);
  float* CLU  = (float*)(ws + OFS_CLU);
  float* TSC  = (float*)(ws + OFS_TSC);
  float* BRP  = (float*)(ws + OFS_BRP);
  unsigned* MAXV = (unsigned*)(ws + OFS_MAXV);
  float* WEFF = (float*)(ws + OFS_WEFF);
  unsigned short* W1T = (unsigned short*)(ws + OFS_W1T);
  unsigned short* W2T = (unsigned short*)(ws + OFS_W2T);
  unsigned short* W3T = (unsigned short*)(ws + OFS_W3T);
  unsigned short* R1T = (unsigned short*)(ws + OFS_R1T);
  unsigned short* R2T = (unsigned short*)(ws + OFS_R2T);
  unsigned short* R3T = (unsigned short*)(ws + OFS_R3T);
  float* OUT = (float*)d_out;

  // ---- init ----
  long long nAdj2 = 2LL * G_ * NPG_ * NPG_;
  k_zero_f<<<gcnt(nAdj2), 256, 0, stream>>>(ADJ, nAdj2);                 // adj+wadj
  k_zero_f<<<gcnt(3LL * NNODES), 256, 0, stream>>>(DEG, 3LL * NNODES);   // deg,wdeg,tri
  k_zero_f<<<gcnt((long long)NNODES * (1 + EDGEH_)), 256, 0, stream>>>(CNT, (long long)NNODES * (1 + EDGEH_));
  k_zero_u<<<1, 256, 0, stream>>>(MAXV, 64);
  k_zero_f<<<1, 32, 0, stream>>>(OUT + OUT_AUX, 1);

  // ---- structure features ----
  k_deg<<<gcnt(NEDGES), 256, 0, stream>>>(src, EATTR, DEG, WDEG);
  k_adjbuild<<<gcnt(NEDGES), 256, 0, stream>>>(src, dst, EATTR, ADJ, WADJ);
  k_adjh<<<gcnt((long long)G_ * NPG_ * NPG_), 256, 0, stream>>>(ADJ, ADJH);
  k_adjtri<<<dim3(16, 4, G_), 128, 0, stream>>>(ADJH, TRI);              // WMMA bf16 (exact 0/1)
  k_rowsum<<<G_, NPG_, 0, stream>>>(ADJH, WADJ, LDEG, WROW);
  k_maxdw<<<gcnt(NNODES), 256, 0, stream>>>(DEG, WDEG, MAXV);
  k_structfin<<<G_, NPG_, 0, stream>>>(LDEG, TRI, WROW, CLU, TSC, BRP, MAXV, OUT);

  // ---- weight transposes (f32 -> bf16, B^T for contiguous WMMA B-frags) ----
  k_wt<<<gcnt(HID_ * HC_), 256, 0, stream>>>(P(19), W1T, HID_, HC_);  // gat1.lin_w 64x256
  k_wt<<<gcnt(HC_ * HC_), 256, 0, stream>>>(P(25), W2T, HC_, HC_);    // gat2.lin_w 256x256
  k_wt<<<gcnt(HC_ * HID_), 256, 0, stream>>>(P(31), W3T, HC_, HID_);  // gat3.lin_w 256x64
  k_wt<<<gcnt(HID_ * HC_), 256, 0, stream>>>(P(45), R1T, HID_, HC_);  // res1.w
  k_wt<<<gcnt(HC_ * HC_), 256, 0, stream>>>(P(47), R2T, HC_, HC_);    // res2.w
  k_wt<<<gcnt(HC_ * HID_), 256, 0, stream>>>(P(49), R3T, HC_, HID_);  // res3.w

  // ---- node projection + edge encoder ----
  k_nodeproj<<<gcnt((long long)NNODES * HID_), 256, 0, stream>>>(
      X, DEG, WDEG, CLU, TSC, BRP, MAXV, P(37), P(36), X0H);
  k_edgeenc<<<gcnt(NEDGES), 256, 0, stream>>>(EATTR, P(7), P(6), P(9), P(8), EAA);
  k_loopacc<<<gcnt(NEDGES), 256, 0, stream>>>(dst, EAA, CNT, LOOP);
  k_loopwrite<<<gcnt((long long)NNODES * EDGEH_), 256, 0, stream>>>(CNT, LOOP, EAA);

  // ---- GAT layer 1 (x0h -> h1h) ----
  k_gemm<<<dim3(HC_ / 16, NNODES / 64), 128, 0, stream>>>(X0H, W1T, nullptr, HX, NNODES, HC_, HID_, 0);
  k_gemm<<<dim3(HC_ / 16, NNODES / 64), 128, 0, stream>>>(X0H, R1T, P(44), ACC, NNODES, HC_, HID_, 0);
  k_weff<<<1, 64, 0, stream>>>(P(18), P(15), WEFF, HEADS_);
  k_srcdst<<<gcnt((long long)NNODES * HEADS_), 256, 0, stream>>>(HX, P(16), P(14), ASRC, ADST, HEADS_);
  k_zero_u<<<gcnt((long long)NNODES * HEADS_), 256, 0, stream>>>(AMAX, (long long)NNODES * HEADS_);
  k_zero_f<<<gcnt((long long)NNODES * HEADS_), 256, 0, stream>>>(ASUM, (long long)NNODES * HEADS_);
  k_attA<<<gcnt((long long)ETOT * HEADS_), 256, 0, stream>>>(src, dst, EAA, WEFF, ASRC, ADST, ASC, AMAX, HEADS_);
  k_attB<<<gcnt((long long)ETOT * HEADS_), 256, 0, stream>>>(dst, ASC, AMAX, ASUM, HEADS_);
  k_attC<<<gcnt((long long)ETOT * HEADS_), 256, 0, stream>>>(src, dst, ASC, ASUM, HX, ACC, HEADS_);
  k_ln<<<NNODES / 4, 128, 0, stream>>>(ACC, P(17), P(39), P(38), H1H, nullptr, HC_, NNODES);

  // ---- GAT layer 2 (h1h -> h2h) ----
  k_gemm<<<dim3(HC_ / 16, NNODES / 64), 128, 0, stream>>>(H1H, W2T, nullptr, HX, NNODES, HC_, HC_, 0);
  k_gemm<<<dim3(HC_ / 16, NNODES / 64), 128, 0, stream>>>(H1H, R2T, P(46), ACC, NNODES, HC_, HC_, 0);
  k_weff<<<1, 64, 0, stream>>>(P(24), P(21), WEFF, HEADS_);
  k_srcdst<<<gcnt((long long)NNODES * HEADS_), 256, 0, stream>>>(HX, P(22), P(20), ASRC, ADST, HEADS_);
  k_zero_u<<<gcnt((long long)NNODES * HEADS_), 256, 0, stream>>>(AMAX, (long long)NNODES * HEADS_);
  k_zero_f<<<gcnt((long long)NNODES * HEADS_), 256, 0, stream>>>(ASUM, (long long)NNODES * HEADS_);
  k_attA<<<gcnt((long long)ETOT * HEADS_), 256, 0, stream>>>(src, dst, EAA, WEFF, ASRC, ADST, ASC, AMAX, HEADS_);
  k_attB<<<gcnt((long long)ETOT * HEADS_), 256, 0, stream>>>(dst, ASC, AMAX, ASUM, HEADS_);
  k_attC<<<gcnt((long long)ETOT * HEADS_), 256, 0, stream>>>(src, dst, ASC, ASUM, HX, ACC, HEADS_);
  k_ln<<<NNODES / 4, 128, 0, stream>>>(ACC, P(23), P(41), P(40), H2H, nullptr, HC_, NNODES);

  // ---- GAT layer 3 (h2h -> h3, single head) ----
  k_gemm<<<dim3(HID_ / 16, NNODES / 64), 128, 0, stream>>>(H2H, W3T, nullptr, HX, NNODES, HID_, HC_, 0);
  k_gemm<<<dim3(HID_ / 16, NNODES / 64), 128, 0, stream>>>(H2H, R3T, P(48), ACC, NNODES, HID_, HC_, 0);
  k_weff<<<1, 64, 0, stream>>>(P(30), P(27), WEFF, 1);
  k_srcdst<<<gcnt(NNODES), 256, 0, stream>>>(HX, P(28), P(26), ASRC, ADST, 1);
  k_zero_u<<<gcnt(NNODES), 256, 0, stream>>>(AMAX, NNODES);
  k_zero_f<<<gcnt(NNODES), 256, 0, stream>>>(ASUM, NNODES);
  k_attA<<<gcnt(ETOT), 256, 0, stream>>>(src, dst, EAA, WEFF, ASRC, ADST, ASC, AMAX, 1);
  k_attB<<<gcnt(ETOT), 256, 0, stream>>>(dst, ASC, AMAX, ASUM, 1);
  k_attC<<<gcnt(ETOT), 256, 0, stream>>>(src, dst, ASC, ASUM, HX, ACC, 1);
  k_ln<<<NNODES / 4, 128, 0, stream>>>(ACC, P(29), P(43), P(42), nullptr, H3, HID_, NNODES);

  // ---- pooling + heads ----
  k_pool<<<G_, NPG_, 0, stream>>>(H3, P(33), P(32), P(35), P(34), OUT);
  k_fus<<<G_, 64, 0, stream>>>(CLINI, METAB, P(11), P(10), P(13), P(12), P(51), P(50), OUT);
  k_shaux<<<G_, 32, 0, stream>>>(P(53), P(52), P(55), P(54), OUT);
}
